// SimilarityModel_24584392803023
// MI455X (gfx1250) — matrix operand order repaired and checked by reading, not compiled, not run on hardware
//
#include <hip/hip_runtime.h>

// ---------------------------------------------------------------------------
// SimilarityModel on MI455X (gfx1250, wave32, WMMA).
// out[b,c,q] = leaky( w2 . leaky( TH[b,c,:] + TU[b,q,:] + (u[b,q,:]*h[b,c,:]) @ Whu^T ) + b2 )
// Dominant GEMM done with v_wmma_f32_16x16x32_bf16; W_hu staged into LDS with
// GLOBAL_LOAD_ASYNC_TO_LDS_B128 (ASYNCcnt), consumed via conflict-free ds_load_b128.
// ---------------------------------------------------------------------------

typedef __attribute__((ext_vector_type(16))) __bf16 v16bf;
typedef __attribute__((ext_vector_type(8)))  __bf16 v8bf;
typedef __attribute__((ext_vector_type(8)))  float  v8f;

#define BB   4
#define CC   384
#define QQ   384
#define HH   256
#define KK   256
#define W1LD 768        // w1 row stride (3*H)

#define LDS_PITCH 264   // bf16 elems per LDS row of Whu: 132 dwords, 132%64==4 -> conflict-free
#define SMEM_BYTES (256 * LDS_PITCH * 2 + 3 * 256 * 4)   // Whu + s + TH + w2 = 138240

// ---------------------------------------------------------------------------
// Prep 1: convert W_hu (= w1[:, 2H:3H]) to bf16 row-major [k][h]
// ---------------------------------------------------------------------------
__global__ void prep_whu_bf16(const float* __restrict__ w1, __bf16* __restrict__ whu) {
    int idx = blockIdx.x * blockDim.x + threadIdx.x;   // 65536 total
    int k = idx >> 8;
    int h = idx & 255;
    whu[idx] = (__bf16)(w1[(size_t)k * W1LD + 2 * HH + h]);
}

// ---------------------------------------------------------------------------
// Prep 2: TH[bc,k] = b1[k] + h[bc,:] . w1[k, 0:H]
//         TU[bq,k] =         u[bq,:] . w1[k, H:2H]
// grid = (1536, 2), block = 256 (thread = k). Tiny GEMV work (0.4 GFLOP).
// ---------------------------------------------------------------------------
__global__ void prep_thtu(const float* __restrict__ hbuf, const float* __restrict__ ubuf,
                          const float* __restrict__ w1,   const float* __restrict__ b1,
                          float* __restrict__ TH, float* __restrict__ TU) {
    __shared__ float row[HH];
    const int r     = blockIdx.x;   // 0..1535 : (b,c) or (b,q)
    const int which = blockIdx.y;   // 0 -> TH, 1 -> TU
    const float* src = which ? (ubuf + (size_t)r * HH) : (hbuf + (size_t)r * HH);
    row[threadIdx.x] = src[threadIdx.x];
    __syncthreads();
    const int k = threadIdx.x;
    const float* wrow = w1 + (size_t)k * W1LD + (which ? HH : 0);
    float acc = which ? 0.0f : b1[k];
#pragma unroll 8
    for (int j = 0; j < HH; ++j) acc = fmaf(row[j], wrow[j], acc);
    (which ? TU : TH)[(size_t)r * KK + k] = acc;
}

__device__ __forceinline__ float leaky(float v) {
    return fmaxf(v, 0.0f) + 0.01f * fminf(v, 0.0f);
}

// ---------------------------------------------------------------------------
// Main kernel: one workgroup (256 thr = 8 wave32) per (b,c).
// Per (b,c): GEMM  M=384 (q) x N=256 (k) x K=256 (h) in bf16 WMMA, epilogue fused.
// ---------------------------------------------------------------------------
__global__ __launch_bounds__(256, 1)
void simmodel_main(const float* __restrict__ hbuf, const float* __restrict__ ubuf,
                   const __bf16* __restrict__ whu,
                   const float* __restrict__ TH, const float* __restrict__ TU,
                   const float* __restrict__ w2, const float* __restrict__ b2,
                   float* __restrict__ out) {
    extern __shared__ char smem[];
    __bf16* whuL = (__bf16*)smem;                            // [256][LDS_PITCH]
    float*  sL   = (float*)(smem + 256 * LDS_PITCH * 2);     // h[b,c,:] (256)
    float*  thL  = sL + HH;                                  // TH[b,c,:] (256)
    float*  w2L  = thL + KK;                                 // w2 (256)

    const int tid = threadIdx.x;
    const int bc  = blockIdx.x;              // b*C + c
    const int b   = bc / CC;

    // ---- stage Whu bf16 -> LDS via async copy (no VGPR round-trip) ----
    // Each lane: 16B chunk; global byte offset = idx*16, LDS offset = padded row layout.
    {
        const unsigned long long gbase = (unsigned long long)(uintptr_t)whu;
#pragma unroll 4
        for (int idx = tid; idx < 8192; idx += 256) {
            const int row = idx >> 5;            // 32 chunks of 8 bf16 per 256-col row
            const int c8  = idx & 31;
            const unsigned int goff   = (unsigned int)(idx << 4);   // row*512 + c8*16
            const unsigned int ldsoff =
                (unsigned int)(uintptr_t)(void*)(whuL + row * LDS_PITCH + c8 * 8);
            asm volatile("global_load_async_to_lds_b128 %0, %1, %2"
                         :
                         : "v"(ldsoff), "v"(goff), "s"(gbase)
                         : "memory");
        }
    }

    // stage per-(b,c) vectors (overlaps with async DMA above)
    sL[tid]  = hbuf[(size_t)bc * HH + tid];
    thL[tid] = TH[(size_t)bc * KK + tid];
    w2L[tid] = w2[tid];

    // drain this wave's async copies, then barrier so all waves see full LDS image
    asm volatile("s_wait_asynccnt 0x0" ::: "memory");
    __syncthreads();

    const int wave = tid >> 5;
    const int lane = tid & 31;
    const int mrow = lane & 15;              // A: M index / B: N index / C: N index
    const int hi   = (lane >> 4) & 1;        // lane half selects K sub-range
    const int koff = hi ? 8 : 0;             // A-matrix K offset for this lane half
    const float bias2 = b2[0];

    // 24 M-tiles of q, 3 per wave
    for (int mt = wave; mt < 24; mt += 8) {
        const int q0 = mt * 16;
        const float* urow = ubuf + ((size_t)(b * QQ) + q0 + mrow) * HH;

        // --- build A fragments: A[q,h] = u[b,q,h] * h[b,c,h], bf16, ISA 16x32 layout ---
        v16bf afr[8];
#pragma unroll
        for (int ht = 0; ht < 8; ++ht) {
            const int h0 = ht * 32;
            const float4 u0 = *(const float4*)(urow + h0 + koff);
            const float4 u1 = *(const float4*)(urow + h0 + koff + 4);
            const float4 u2 = *(const float4*)(urow + h0 + 16 + koff);
            const float4 u3 = *(const float4*)(urow + h0 + 16 + koff + 4);
            const float* sp = sL + h0 + koff;
            v16bf af;
            af[0]  = (__bf16)(u0.x * sp[0]);  af[1]  = (__bf16)(u0.y * sp[1]);
            af[2]  = (__bf16)(u0.z * sp[2]);  af[3]  = (__bf16)(u0.w * sp[3]);
            af[4]  = (__bf16)(u1.x * sp[4]);  af[5]  = (__bf16)(u1.y * sp[5]);
            af[6]  = (__bf16)(u1.z * sp[6]);  af[7]  = (__bf16)(u1.w * sp[7]);
            af[8]  = (__bf16)(u2.x * sp[16]); af[9]  = (__bf16)(u2.y * sp[17]);
            af[10] = (__bf16)(u2.z * sp[18]); af[11] = (__bf16)(u2.w * sp[19]);
            af[12] = (__bf16)(u3.x * sp[20]); af[13] = (__bf16)(u3.y * sp[21]);
            af[14] = (__bf16)(u3.z * sp[22]); af[15] = (__bf16)(u3.w * sp[23]);
            afr[ht] = af;
        }

        float acc[8] = {0.f, 0.f, 0.f, 0.f, 0.f, 0.f, 0.f, 0.f};

        for (int nt = 0; nt < 16; ++nt) {
            const int k0   = nt * 16;
            const int kcol = k0 + mrow;      // this lane's N (=k) column

            // init C = TH[k] + TU[q,k]  (b1 folded into TH)
            v8f cacc;
            const float thv = thL[kcol];
#pragma unroll
            for (int r = 0; r < 8; ++r) {
                const int q = q0 + r + (hi ? 8 : 0);
                cacc[r] = thv + TU[((size_t)(b * QQ) + q) * KK + kcol];
            }

            // K loop: 8 x v_wmma_f32_16x16x32_bf16
#pragma unroll
            for (int ht = 0; ht < 8; ++ht) {
                const int h0 = ht * 32;
                const __bf16* bp = whuL + (size_t)kcol * LDS_PITCH + h0 + (hi ? 8 : 0);
                const v8bf blo = *(const v8bf*)(bp);       // 16B ds load, conflict-free
                const v8bf bhi = *(const v8bf*)(bp + 16);
                const v16bf bfr = __builtin_shufflevector(
                    blo, bhi, 0, 1, 2, 3, 4, 5, 6, 7, 8, 9, 10, 11, 12, 13, 14, 15);
                cacc = __builtin_amdgcn_wmma_f32_16x16x32_bf16(
                    false, afr[ht], false, bfr, (short)0, cacc, false, false);
            }

            // epilogue: leaky + dot with w2 (partial per lane over its k column)
            const float wk = w2L[kcol];
#pragma unroll
            for (int r = 0; r < 8; ++r) {
                acc[r] = fmaf(wk, leaky(cacc[r]), acc[r]);
            }
        }

        // reduce over the 16 lanes sharing each q row (lanes 0-15 / 16-31 groups)
#pragma unroll
        for (int r = 0; r < 8; ++r) {
            float v = acc[r];
            v += __shfl_xor(v, 1, 32);
            v += __shfl_xor(v, 2, 32);
            v += __shfl_xor(v, 4, 32);
            v += __shfl_xor(v, 8, 32);
            acc[r] = v;
        }
        if (mrow == 0) {
            float* orow = out + (size_t)bc * QQ + q0 + (hi ? 8 : 0);
#pragma unroll
            for (int r = 0; r < 8; ++r) {
                orow[r] = leaky(acc[r] + bias2);
            }
        }
    }
}

// ---------------------------------------------------------------------------
// Launcher
// ---------------------------------------------------------------------------
extern "C" void kernel_launch(void* const* d_in, const int* in_sizes, int n_in,
                              void* d_out, int out_size, void* d_ws, size_t ws_size,
                              hipStream_t stream) {
    const float* h  = (const float*)d_in[0];
    const float* u  = (const float*)d_in[1];
    const float* w1 = (const float*)d_in[2];
    const float* b1 = (const float*)d_in[3];
    const float* w2 = (const float*)d_in[4];
    const float* b2 = (const float*)d_in[5];
    float* out = (float*)d_out;

    char* ws = (char*)d_ws;
    __bf16* whu = (__bf16*)ws;                                  // 131072 B
    float*  TH  = (float*)(ws + 131072);                        // 1536*256*4
    float*  TU  = (float*)(ws + 131072 + 1536 * 256 * 4);       // 1536*256*4

    prep_whu_bf16<<<256, 256, 0, stream>>>(w1, whu);
    prep_thtu<<<dim3(1536, 2), 256, 0, stream>>>(h, u, w1, b1, TH, TU);

    // 138 KB dynamic LDS (gfx1250 WGP has 320 KB); opt in above the default cap.
    (void)hipFuncSetAttribute((const void*)simmodel_main,
                              hipFuncAttributeMaxDynamicSharedMemorySize, SMEM_BYTES);
    simmodel_main<<<BB * CC, 256, SMEM_BYTES, stream>>>(h, u, whu, TH, TU, w2, b2, out);
}
